// GRU_249108103354
// MI455X (gfx1250) — compile-verified, matrix operand order checked by reading, not run on hardware
//
#include <hip/hip_runtime.h>

// ---------------------------------------------------------------------------
// GRU on MI455X (gfx1250, wave32, WMMA + Tensor Data Mover):
//   phase 0: f32->bf16 conversion + W_hh repack (per-WG LDS-resident slices)
//   phase 1: gx = X @ W_ih^T + b_ih   (bf16 WMMA GEMM, TDM-staged LDS tiles,
//                                      double-buffered via TENSORcnt)
//   phase 2: persistent 64-WG recurrence, W_hh pinned in LDS for all 16384
//            steps (one 96KB TDM load with HW padding), per-step h staged by
//            a 2KB TDM load, matvec via v_wmma_f32_16x16x32_bf16,
//            device-scope arrive-and-spin barrier (ping-pong h buffers)
//   phase 3: out = hs @ W_fc^T + b_fc (bf16 WMMA GEMM, f32 out)
// ---------------------------------------------------------------------------

#define T_SEQ  16384
#define D_IN   512
#define D_H    1024
#define D_G3   3072
#define D_OUT  512

#define NWG_REC     64
#define H_PER_WG    16            // h elements owned per WG
#define ROWS_PER_WG 48            // 3 gates x 16 rows
#define LDS_STRIDE  1032          // 1024 + 8 pad halfwords (2064B rows)

#if defined(__has_builtin)
#  if __has_builtin(__builtin_amdgcn_tensor_load_to_lds)
#    define HAVE_TDM 1
#  endif
#endif
#ifndef HAVE_TDM
#  define HAVE_TDM 0
#endif

typedef __attribute__((ext_vector_type(16))) __bf16 bf16x16;
typedef __attribute__((ext_vector_type(8)))  float  f32x8;
typedef __attribute__((ext_vector_type(4)))  unsigned int tdm_u32x4;
typedef __attribute__((ext_vector_type(8)))  int          tdm_i32x8;
typedef __attribute__((ext_vector_type(4)))  int          tdm_i32x4;

// ---------------- scalar helpers ----------------

__device__ __forceinline__ unsigned short f2bf(float f) {
  union { float f; unsigned int u; } v; v.f = f;
  unsigned int u = v.u;
  unsigned int r = u + 0x7FFFu + ((u >> 16) & 1u);   // round-to-nearest-even
  return (unsigned short)(r >> 16);
}

__device__ __forceinline__ float bf2f(unsigned short h) {
  union { unsigned int u; float f; } v; v.u = ((unsigned int)h) << 16;
  return v.f;
}

__device__ __forceinline__ float sigmoidf(float x) {
  return 1.0f / (1.0f + __expf(-x));
}

// LDS byte address (address_space(3) offset) of a shared-memory pointer,
// for TDM descriptor construction.
__device__ __forceinline__ unsigned lds_offset_of(const void* p) {
  return (unsigned)(size_t)(const __attribute__((address_space(3))) void*)p;
}

#if HAVE_TDM
// Issue one 2D TENSOR_LOAD_TO_LDS. Descriptor packed per CDNA5 D# spec:
//  group0: count=1 | lds_addr | global_addr[56:0] | type=2
//  group1: data_size=1(2B), pad ctl, tensor_dim0/1 = tile_w/h (tile always
//          in-bounds), tile_dim0/1, tensor_dim0_stride (elements)
// pad_interval: bytesStored/8 >= 2^pi -> insert pad; pad_amount: (pa+1) DWORDs.
__device__ __forceinline__ void tdm_load_2d(unsigned lds_addr, const void* gptr,
                                            unsigned tile_w, unsigned tile_h,
                                            unsigned stride_elems,
                                            unsigned pad_en, unsigned pad_interval,
                                            unsigned pad_amount) {
  unsigned long long ga = (unsigned long long)(size_t)gptr;
  tdm_u32x4 g0 = { 1u,                                  // count=1 (valid, user)
                   lds_addr,
                   (unsigned)ga,
                   ((unsigned)((ga >> 32) & 0x01ffffffu)) | (2u << 30) };
  tdm_i32x8 g1 = { 0, 0, 0, 0, 0, 0, 0, 0 };
  g1[0] = (int)((1u << 16) | (pad_en << 20) |
                (pad_interval << 22) | (pad_amount << 25));
  g1[1] = (int)((tile_w & 0xffffu) << 16);              // tensor_dim0 lo
  g1[2] = (int)((tile_h & 0xffffu) << 16);              // tensor_dim1 lo
  g1[3] = (int)((tile_w & 0xffffu) << 16);              // tile_dim0
  g1[4] = (int)(tile_h & 0xffffu);                      // tile_dim1
  g1[5] = (int)stride_elems;                            // dim0 stride lo
  tdm_i32x4 gz = { 0, 0, 0, 0 };
#if __clang_major__ >= 23
  tdm_i32x8 gz8 = { 0, 0, 0, 0, 0, 0, 0, 0 };
  __builtin_amdgcn_tensor_load_to_lds(g0, g1, gz, gz, gz8, 0);
#else
  __builtin_amdgcn_tensor_load_to_lds(g0, g1, gz, gz, 0);
#endif
}
#endif  // HAVE_TDM

// Load one 16x32 bf16 WMMA fragment from an LDS tile laid out row-major
// (row stride in halfwords). Per ISA 7.12.2 (16-bit A/B 16x32):
//   lanes 0-15 : row = lane,    K = {0..7, 16..23}
//   lanes 16-31: row = lane-16, K = {8..15, 24..31}
// stride==0 broadcasts one 32-element vector into all rows/columns
// (used to feed h as the B matrix of the recurrence matvec).
__device__ __forceinline__ bf16x16 load_frag(const unsigned short* p,
                                             int stride, int lane) {
  const int r   = lane & 15;
  const int off = (lane < 16) ? 0 : 8;
  const unsigned short* q = p + r * stride + off;
  union { bf16x16 v; uint4 u[2]; } t;
  t.u[0] = *(const uint4*)(q);        // K block 0
  t.u[1] = *(const uint4*)(q + 16);   // K block +16
  return t.v;
}

// ---------------- phase 0 kernels ----------------

__global__ void k_cvt_bf16(const float* __restrict__ in,
                           unsigned short* __restrict__ out, int n) {
  int i = blockIdx.x * blockDim.x + threadIdx.x;
  if (i < n) out[i] = f2bf(in[i]);
}

// Repack W_hh (row-major [3072][1024] f32) into per-WG contiguous bf16 slices:
// slice wg holds rows { g*1024 + wg*16 + m : g in 0..2, m in 0..15 }.
__global__ void k_pack_whh(const float* __restrict__ W,
                           unsigned short* __restrict__ out, int n) {
  int i = blockIdx.x * blockDim.x + threadIdx.x;
  if (i >= n) return;
  int k  = i & (D_H - 1);
  int ro = i >> 10;                       // packed row index
  int wg = ro / ROWS_PER_WG;
  int lr = ro % ROWS_PER_WG;
  int grow = (lr >> 4) * D_H + wg * H_PER_WG + (lr & 15);
  out[i] = f2bf(W[(size_t)grow * D_H + k]);
}

__global__ void k_init_state(float* hA, float* hB, unsigned int* ctr) {
  for (int j = threadIdx.x; j < D_H; j += blockDim.x) { hA[j] = 0.f; hB[j] = 0.f; }
  if (threadIdx.x == 0) *ctr = 0u;
}

// ---------------- phases 1 & 3: tiled WMMA GEMM ----------------
// C[M,N] = Arm[M,K] * Brm[N,K]^T + bias[N]; A,B bf16 row-major (K inner).
// Block = 256 threads = 8 waves; block tile 128x128; wave tile 32x64
// (2x4 WMMA 16x16 tiles). K staged 32-wide through LDS by the Tensor Data
// Mover with hardware padding (16B pad per 64B row -> stride 40 halfwords,
// conflict-free fragment reads), double-buffered on TENSORcnt.

template <bool OUT_BF16>
__launch_bounds__(256)
__global__ void k_gemm_bf16_wmma(const unsigned short* __restrict__ A, int lda,
                                 const unsigned short* __restrict__ B, int ldb,
                                 const float* __restrict__ bias,
                                 void* __restrict__ Cptr, int ldc, int K) {
  const int tid  = threadIdx.x;
  const int row0 = blockIdx.x * 128;
  const int col0 = blockIdx.y * 128;
  const int w    = tid >> 5;
  const int lane = tid & 31;
  const int Mb   = (w & 3) * 32;            // wave M offset in block tile
  const int Nb   = (w >> 2) * 64;           // wave N offset in block tile

  f32x8 acc[2][4];
  for (int mt = 0; mt < 2; ++mt)
    for (int nt = 0; nt < 4; ++nt)
      acc[mt][nt] = (f32x8){};

  const int nkc = K >> 5;

#if HAVE_TDM
  __shared__ unsigned short As[2][128 * 40];
  __shared__ unsigned short Bs[2][128 * 40];
  const unsigned as_off[2] = { lds_offset_of(&As[0][0]), lds_offset_of(&As[1][0]) };
  const unsigned bs_off[2] = { lds_offset_of(&Bs[0][0]), lds_offset_of(&Bs[1][0]) };

  if (w == 0) {  // prologue: stage k-chunk 0 into buffer 0
    tdm_load_2d(as_off[0], &A[(size_t)row0 * lda], 32, 128, (unsigned)lda, 1, 3, 3);
    tdm_load_2d(bs_off[0], &B[(size_t)col0 * ldb], 32, 128, (unsigned)ldb, 1, 3, 3);
  }

  for (int kc = 0; kc < nkc; ++kc) {
    const int cur = kc & 1;
    __syncthreads();   // all waves done computing on the buffer we overwrite
    if (w == 0) {
      if (kc + 1 < nkc) {
        const int k1 = (kc + 1) << 5;
        tdm_load_2d(as_off[cur ^ 1], &A[(size_t)row0 * lda + k1], 32, 128,
                    (unsigned)lda, 1, 3, 3);
        tdm_load_2d(bs_off[cur ^ 1], &B[(size_t)col0 * ldb + k1], 32, 128,
                    (unsigned)ldb, 1, 3, 3);
        __builtin_amdgcn_s_wait_tensorcnt(2);  // pair(kc) complete, pair(kc+1) in flight
      } else {
        __builtin_amdgcn_s_wait_tensorcnt(0);
      }
    }
    __syncthreads();

    bf16x16 afrag[2], bfrag[4];
    for (int mt = 0; mt < 2; ++mt)
      afrag[mt] = load_frag(&As[cur][(Mb + mt * 16) * 40], 40, lane);
    for (int nt = 0; nt < 4; ++nt)
      bfrag[nt] = load_frag(&Bs[cur][(Nb + nt * 16) * 40], 40, lane);

    for (int mt = 0; mt < 2; ++mt)
      for (int nt = 0; nt < 4; ++nt)
        acc[mt][nt] = __builtin_amdgcn_wmma_f32_16x16x32_bf16(
            false, afrag[mt], false, bfrag[nt], (short)0, acc[mt][nt],
            false, false);
  }
#else
  __shared__ unsigned short As[128 * 40];
  __shared__ unsigned short Bs[128 * 40];
  for (int kc = 0; kc < nkc; ++kc) {
    const int k0 = kc << 5;
    __syncthreads();
    for (int j = 0; j < 2; ++j) {
      int ci   = tid + (j << 8);
      int rrow = ci >> 2;
      int c8   = (ci & 3) << 3;
      *(uint4*)&As[rrow * 40 + c8] =
          *(const uint4*)&A[(size_t)(row0 + rrow) * lda + k0 + c8];
      *(uint4*)&Bs[rrow * 40 + c8] =
          *(const uint4*)&B[(size_t)(col0 + rrow) * ldb + k0 + c8];
    }
    __syncthreads();
    bf16x16 afrag[2], bfrag[4];
    for (int mt = 0; mt < 2; ++mt)
      afrag[mt] = load_frag(&As[(Mb + mt * 16) * 40], 40, lane);
    for (int nt = 0; nt < 4; ++nt)
      bfrag[nt] = load_frag(&Bs[(Nb + nt * 16) * 40], 40, lane);
    for (int mt = 0; mt < 2; ++mt)
      for (int nt = 0; nt < 4; ++nt)
        acc[mt][nt] = __builtin_amdgcn_wmma_f32_16x16x32_bf16(
            false, afrag[mt], false, bfrag[nt], (short)0, acc[mt][nt],
            false, false);
  }
#endif

  // D layout: lane n holds column (n&15), rows r + (lane<16 ? 0 : 8)
  for (int mt = 0; mt < 2; ++mt) {
    int mbase = row0 + Mb + mt * 16 + ((lane < 16) ? 0 : 8);
    for (int nt = 0; nt < 4; ++nt) {
      int col  = col0 + Nb + nt * 16 + (lane & 15);
      float bv = bias[col];
      for (int r = 0; r < 8; ++r) {
        float v = acc[mt][nt][r] + bv;
        if (OUT_BF16)
          ((unsigned short*)Cptr)[(size_t)(mbase + r) * ldc + col] = f2bf(v);
        else
          ((float*)Cptr)[(size_t)(mbase + r) * ldc + col] = v;
      }
    }
  }
}

// ---------------- phase 2: persistent GRU recurrence ----------------
// 64 WGs x 128 threads. Each WG pins its 48x1024 bf16 W_hh slice in LDS for
// all 16384 steps (one TDM load; HW pad of 8B per 1024B => 2064B row stride,
// conflict-free). Per step: wave 0 TDM-loads h(t)=hs_bf[t-1] (2KB) into LDS,
// waves 0..2 each compute one 16-row gate block of gh = W_hh @ h via 32
// chained WMMAs; threads 0..15 apply the gates for the WG's 16 owned h
// elements; ping-pong f32 h buffers + device-scope spin barrier.

__launch_bounds__(128)
__global__ void k_gru_recurrence(const unsigned short* __restrict__ whh_pk,
                                 const unsigned short* __restrict__ gx,
                                 const float* __restrict__ b_hh,
                                 float* __restrict__ hA, float* __restrict__ hB,
                                 unsigned short* __restrict__ hs_bf,
                                 unsigned int* __restrict__ ctr, int T) {
  extern __shared__ unsigned char smem_raw[];
  unsigned short* whh_s = (unsigned short*)smem_raw;           // 48 x 1032
  unsigned short* h_s   = whh_s + ROWS_PER_WG * LDS_STRIDE;    // 1024 bf16
  float*          gh_s  = (float*)(h_s + D_H);                 // 48 f32

  const int wg   = blockIdx.x;
  const int tid  = threadIdx.x;
  const int w    = tid >> 5;
  const int lane = tid & 31;

  // One-time load: W_hh slice -> LDS (resident for all T steps).
  // LDS element (row, k) lives at byte row*2064 + k*2 + (k>=512 ? 16 : 0)
  // (16B pad after the first 1024B of each row, none at the row end).
  {
#if HAVE_TDM
    if (w == 0) {
      tdm_load_2d(lds_offset_of(whh_s),
                  whh_pk + (size_t)wg * ROWS_PER_WG * D_H,
                  D_H, ROWS_PER_WG, D_H, 1, 7, 1);
      __builtin_amdgcn_s_wait_tensorcnt(0);
    }
#else
    const unsigned short* src = whh_pk + (size_t)wg * ROWS_PER_WG * D_H;
    for (int c = tid; c < (ROWS_PER_WG * D_H) / 8; c += blockDim.x) {
      int row = c >> 7;
      int c8  = (c & 127) << 3;
      *(uint4*)&whh_s[row * LDS_STRIDE + c8 + (c8 >= 512 ? 8 : 0)] =
          *(const uint4*)&src[(size_t)row * D_H + c8];
    }
#endif
    for (int i = tid; i < D_H; i += blockDim.x) h_s[i] = 0;  // h(0) = 0
  }
  __syncthreads();

  for (int t = 0; t < T; ++t) {
    const float* hc = (t & 1) ? hB : hA;
    float*       hn = (t & 1) ? hA : hB;

    // Stage h(t) (bf16) into LDS: h(t) is row t-1 of hs_bf.
#if HAVE_TDM
    if (t > 0 && w == 0) {
      tdm_load_2d(lds_offset_of(h_s), hs_bf + (size_t)(t - 1) * D_H,
                  D_H, 1, D_H, 0, 0, 0);
      __builtin_amdgcn_s_wait_tensorcnt(0);
    }
#else
    for (int i = tid; i < D_H; i += blockDim.x) h_s[i] = f2bf(hc[i]);
#endif
    __syncthreads();

    if (w < 3) {  // wave w computes gate block w (16 rows x 1024 K)
      f32x8 acc = (f32x8){};
      const unsigned short* ap = whh_s + w * 16 * LDS_STRIDE;
      for (int kc = 0; kc < D_H / 32; ++kc) {
        const int koff = kc * 32 + (kc >= 16 ? 8 : 0);   // skip mid-row pad
        bf16x16 a = load_frag(ap + koff, LDS_STRIDE, lane);
        bf16x16 b = load_frag(h_s + kc * 32, 0, lane);   // broadcast columns
        acc = __builtin_amdgcn_wmma_f32_16x16x32_bf16(
            false, a, false, b, (short)0, acc, false, false);
      }
      // Matvec result = column 0 of D: lanes 0 (rows 0..7) and 16 (rows 8..15).
      if (lane == 0)
        for (int r = 0; r < 8; ++r) gh_s[w * 16 + r] = acc[r];
      if (lane == 16)
        for (int r = 0; r < 8; ++r) gh_s[w * 16 + 8 + r] = acc[r];
    }
    __syncthreads();

    if (tid < H_PER_WG) {
      int gi = wg * H_PER_WG + tid;
      float ghr = gh_s[tid]      + b_hh[gi];
      float ghz = gh_s[16 + tid] + b_hh[D_H + gi];
      float ghn = gh_s[32 + tid] + b_hh[2 * D_H + gi];
      size_t gb = (size_t)t * D_G3;
      float gxr = bf2f(gx[gb + gi]);
      float gxz = bf2f(gx[gb + D_H + gi]);
      float gxn = bf2f(gx[gb + 2 * D_H + gi]);
      float rr = sigmoidf(gxr + ghr);
      float zz = sigmoidf(gxz + ghz);
      float nn = tanhf(gxn + rr * ghn);
      float hv = (1.0f - zz) * nn + zz * hc[gi];
      hn[gi] = hv;
      hs_bf[(size_t)t * D_H + gi] = f2bf(hv);
    }

    // Device-scope arrive-and-spin barrier (monotonic counter).
    __threadfence();
    __syncthreads();
    if (tid == 0) {
      __hip_atomic_fetch_add(ctr, 1u, __ATOMIC_ACQ_REL, __HIP_MEMORY_SCOPE_AGENT);
      unsigned int target = (unsigned int)(t + 1) * NWG_REC;
      while (__hip_atomic_load(ctr, __ATOMIC_ACQUIRE, __HIP_MEMORY_SCOPE_AGENT)
             < target) {
        __builtin_amdgcn_s_sleep(1);
      }
    }
    __syncthreads();
  }
}

// ---------------- host launcher ----------------

extern "C" void kernel_launch(void* const* d_in, const int* in_sizes, int n_in,
                              void* d_out, int out_size, void* d_ws,
                              size_t ws_size, hipStream_t stream) {
  (void)in_sizes; (void)n_in; (void)out_size; (void)ws_size;

  const float* X    = (const float*)d_in[0];
  const float* W_ih = (const float*)d_in[1];
  const float* W_hh = (const float*)d_in[2];
  const float* b_ih = (const float*)d_in[3];
  const float* b_hh = (const float*)d_in[4];
  const float* W_fc = (const float*)d_in[5];
  const float* b_fc = (const float*)d_in[6];
  float* out = (float*)d_out;

  // Workspace layout (all sizes naturally 256B aligned), ~161.5 MB total.
  char* ws = (char*)d_ws;
  constexpr size_t SZ_XBF = (size_t)T_SEQ * D_IN * 2;   // 16 MB
  constexpr size_t SZ_WIH = (size_t)D_G3 * D_IN * 2;    //  3 MB
  constexpr size_t SZ_WFC = (size_t)D_OUT * D_H * 2;    //  1 MB
  constexpr size_t SZ_WHH = (size_t)D_G3 * D_H * 2;     //  6 MB
  constexpr size_t SZ_GX  = (size_t)T_SEQ * D_G3 * 2;   // 96 MB
  constexpr size_t SZ_HS  = (size_t)T_SEQ * D_H * 2;    // 32 MB
  unsigned short* Xbf    = (unsigned short*)(ws);
  unsigned short* Wih_bf = (unsigned short*)(ws + SZ_XBF);
  unsigned short* Wfc_bf = (unsigned short*)(ws + SZ_XBF + SZ_WIH);
  unsigned short* Whh_pk = (unsigned short*)(ws + SZ_XBF + SZ_WIH + SZ_WFC);
  unsigned short* gx_bf  = (unsigned short*)(ws + SZ_XBF + SZ_WIH + SZ_WFC + SZ_WHH);
  unsigned short* hs_bf  = (unsigned short*)(ws + SZ_XBF + SZ_WIH + SZ_WFC + SZ_WHH + SZ_GX);
  char* tail = ws + SZ_XBF + SZ_WIH + SZ_WFC + SZ_WHH + SZ_GX + SZ_HS;
  float* hA = (float*)(tail);
  float* hB = (float*)(tail + 4096);
  unsigned int* ctr = (unsigned int*)(tail + 8192);

  // Phase 0: conversions + repack + state init.
  {
    int n1 = T_SEQ * D_IN;
    k_cvt_bf16<<<(n1 + 255) / 256, 256, 0, stream>>>(X, Xbf, n1);
    int n2 = D_G3 * D_IN;
    k_cvt_bf16<<<(n2 + 255) / 256, 256, 0, stream>>>(W_ih, Wih_bf, n2);
    int n3 = D_OUT * D_H;
    k_cvt_bf16<<<(n3 + 255) / 256, 256, 0, stream>>>(W_fc, Wfc_bf, n3);
    int n4 = D_G3 * D_H;
    k_pack_whh<<<(n4 + 255) / 256, 256, 0, stream>>>(W_hh, Whh_pk, n4);
    k_init_state<<<1, 256, 0, stream>>>(hA, hB, ctr);
  }

  // Phase 1: gx = X @ W_ih^T + b_ih  (bf16 out).
  {
    dim3 grid(T_SEQ / 128, D_G3 / 128);
    k_gemm_bf16_wmma<true><<<grid, 256, 0, stream>>>(
        Xbf, D_IN, Wih_bf, D_IN, b_ih, (void*)gx_bf, D_G3, D_IN);
  }

  // Phase 2: persistent recurrence (LDS-resident W_hh).
  {
    constexpr int SMEM_REC =
        (ROWS_PER_WG * LDS_STRIDE + D_H) * 2 + ROWS_PER_WG * 4;  // ~99 KB
    hipFuncSetAttribute((const void*)k_gru_recurrence,
                        hipFuncAttributeMaxDynamicSharedMemorySize, SMEM_REC);
    k_gru_recurrence<<<NWG_REC, 128, SMEM_REC, stream>>>(
        Whh_pk, gx_bf, b_hh, hA, hB, hs_bf, ctr, T_SEQ);
  }

  // Phase 3: out = hs @ W_fc^T + b_fc  (f32 out).
  {
    dim3 grid(T_SEQ / 128, D_OUT / 128);
    k_gemm_bf16_wmma<false><<<grid, 256, 0, stream>>>(
        hs_bf, D_H, Wfc_bf, D_H, b_fc, (void*)out, D_OUT, D_H);
  }
}
